// CoxPHLoss_22282290332105
// MI455X (gfx1250) — compile-verified, matrix operand order checked by reading, not run on hardware
//
#include <hip/hip_runtime.h>
#include <math.h>

#define MAX_DUR 100000

typedef __attribute__((ext_vector_type(16))) _Float16 v16h;
typedef __attribute__((ext_vector_type(8)))  float    v8f;

__device__ __forceinline__ unsigned f2ord(float x){
  unsigned u = __float_as_uint(x);
  return (u & 0x80000000u) ? ~u : (u | 0x80000000u);
}
__device__ __forceinline__ float ord2f(unsigned u){
  unsigned b = (u & 0x80000000u) ? (u ^ 0x80000000u) : ~u;
  return __uint_as_float(b);
}
__device__ __forceinline__ int clampd(int d){
  return (d < 0) ? 0 : ((d >= MAX_DUR) ? (MAX_DUR - 1) : d);
}

// ---------------- init workspace (ws is poisoned; must re-init every call) ----
__global__ void cox_init(unsigned long long* __restrict__ A,
                         unsigned long long* __restrict__ B,
                         int* __restrict__ mArr, int* __restrict__ neArr,
                         unsigned* __restrict__ maxbits, unsigned* __restrict__ nEv,
                         unsigned long long* __restrict__ sumLh)
{
  int i = blockIdx.x * blockDim.x + threadIdx.x;
  if (i < MAX_DUR) { A[i] = 0ull; B[i] = 0ull; mArr[i] = -1; neArr[i] = 0; }
  if (i == 0) { *maxbits = 0u; *nEv = 0u; *sumLh = 0ull; }
}

// ---------------- pass 1: max(lh), n_ev, per-bin ne/m, WMMA dot(e, lh) -------
// Full 1024-element wave tiles: unguarded b128 loads, dual WMMA accumulators.
__global__ void cox_pass1(const float* __restrict__ lh, const int* __restrict__ dur,
                          const int* __restrict__ ev, int n,
                          unsigned* __restrict__ maxbits, unsigned* __restrict__ nEv,
                          unsigned long long* __restrict__ sumLh,
                          int* __restrict__ mArr, int* __restrict__ neArr)
{
  const int tid  = blockIdx.x * blockDim.x + threadIdx.x;
  const int lane = threadIdx.x & 31;
  const int wave = tid >> 5;
  const int nW   = (gridDim.x * blockDim.x) >> 5;

  v16h ones;
  #pragma unroll
  for (int k = 0; k < 16; ++k) ones[k] = (_Float16)1.0f;
  v8f c0 = {0.f,0.f,0.f,0.f,0.f,0.f,0.f,0.f};
  v8f c1 = {0.f,0.f,0.f,0.f,0.f,0.f,0.f,0.f};

  float lmax    = -__builtin_inff();
  float tailsum = 0.f;
  int   lcnt    = 0;

  const int fullTiles = n >> 10;   // 1024-element tiles, wave-uniform count

  for (int tile = wave; tile < fullTiles; tile += nW) {
    const int lb = (tile << 10) + lane * 32;   // 32 consecutive elems per lane
    _Float16 aval[32];
    #pragma unroll
    for (int h = 0; h < 8; ++h) {
      const float4 xv  = *(const float4*)(lh  + lb + 4*h);
      const int4   evv = *(const int4*)  (ev  + lb + 4*h);
      const int4   dvv = *(const int4*)  (dur + lb + 4*h);
      const float xs[4] = {xv.x,  xv.y,  xv.z,  xv.w};
      const int   es[4] = {evv.x, evv.y, evv.z, evv.w};
      const int   ds[4] = {dvv.x, dvv.y, dvv.z, dvv.w};
      #pragma unroll
      for (int k = 0; k < 4; ++k) {
        const float x = xs[k];
        const int   e = es[k];
        lmax = fmaxf(lmax, x);
        if (e) {
          const int d = clampd(ds[k]);
          atomicMax(&mArr[d], lb + 4*h + k);   // last event original index
          atomicAdd(&neArr[d], 1);             // events per bin
          ++lcnt;
        }
        aval[4*h + k] = e ? (_Float16)x : (_Float16)0.0f;
      }
    }
    v16h a0, a1;
    #pragma unroll
    for (int k = 0; k < 16; ++k) { a0[k] = aval[k]; a1[k] = aval[16 + k]; }
    // D = A x Ones + C: accumulates row sums of event-masked lh chunks
    c0 = __builtin_amdgcn_wmma_f32_16x16x32_f16(false, a0, false, ones,
                                                (short)0, c0, false, false);
    c1 = __builtin_amdgcn_wmma_f32_16x16x32_f16(false, a1, false, ones,
                                                (short)0, c1, false, false);
  }

  // tail (< 1024 elements): one designated wave, scalar, no WMMA
  if (wave == 0) {
    for (int idx = (fullTiles << 10) + lane; idx < n; idx += 32) {
      const float x = lh[idx];
      lmax = fmaxf(lmax, x);
      if (ev[idx]) {
        const int d = clampd(dur[idx]);
        atomicMax(&mArr[d], idx);
        atomicAdd(&neArr[d], 1);
        ++lcnt;
        tailsum += x;
      }
    }
  }

  // Sum of all 256 D entries == 16 * chunk total (row sums replicated across
  // 16 columns). Per-lane: csum/16 + tail part; wave reduce; fixed-point atomic.
  float s = (c0[0]+c0[1]+c0[2]+c0[3]+c0[4]+c0[5]+c0[6]+c0[7] +
             c1[0]+c1[1]+c1[2]+c1[3]+c1[4]+c1[5]+c1[6]+c1[7]) * (1.0f/16.0f)
            + tailsum;
  #pragma unroll
  for (int off = 16; off > 0; off >>= 1) s += __shfl_xor(s, off, 32);
  if (lane == 0) {
    long long q = (long long)((double)s * 4294967296.0);   // 2^32 fixed point
    atomicAdd(sumLh, (unsigned long long)q);
  }

  // block-level reduce of max and event count -> one atomic per block
  #pragma unroll
  for (int off = 16; off > 0; off >>= 1) {
    lmax = fmaxf(lmax, __shfl_xor(lmax, off, 32));
    lcnt += __shfl_xor(lcnt, off, 32);
  }
  __shared__ float sMx[32];
  __shared__ int   sCt[32];
  const int w = threadIdx.x >> 5;
  if (lane == 0) { sMx[w] = lmax; sCt[w] = lcnt; }
  __syncthreads();
  if (threadIdx.x == 0) {
    const int nw = blockDim.x >> 5;
    float m = sMx[0]; int csum = sCt[0];
    for (int i = 1; i < nw; ++i) { m = fmaxf(m, sMx[i]); csum += sCt[i]; }
    atomicMax(maxbits, f2ord(m));
    atomicAdd(nEv, (unsigned)csum);
  }
}

// ---------------- pass 2: per-bin exp-sums A and B (fixed point u64) ---------
// Durations staged via double-buffered global_load_async_to_lds_b128; each
// thread owns a private LDS slice (writes+reads only its own 4 ints), so no
// workgroup barriers are needed — only per-wave s_wait_asynccnt.
__global__ void cox_pass2(const float* __restrict__ lh, const int* __restrict__ dur,
                          int n, const unsigned* __restrict__ maxbits,
                          const int* __restrict__ mArr,
                          unsigned long long* __restrict__ A,
                          unsigned long long* __restrict__ B)
{
  __shared__ int sdur[2][1024];
  const float M   = ord2f(*maxbits);
  const int   tid = threadIdx.x;
  const int   fullTiles = n >> 10;            // 1024-element tiles
  const float SCALE = 1099511627776.0f;       // 2^40

  const unsigned lds0 = (unsigned)(unsigned long long)(const void*)&sdur[0][tid*4];
  const unsigned lds1 = (unsigned)(unsigned long long)(const void*)&sdur[1][tid*4];

  // prologue: issue first tile into buffer 0
  if ((int)blockIdx.x < fullTiles) {
    const int* gp = dur + ((int)blockIdx.x << 10) + tid*4;
    asm volatile("global_load_async_to_lds_b128 %0, %1, off"
                 :: "v"(lds0), "v"(gp) : "memory");
  }
  int buf = 0;
  for (int tile = blockIdx.x; tile < fullTiles; tile += gridDim.x) {
    const int nxt = tile + gridDim.x;
    if (nxt < fullTiles) {                     // prefetch next tile, other buf
      const int* gp = dur + (nxt << 10) + tid*4;
      const unsigned l = buf ? lds0 : lds1;
      asm volatile("global_load_async_to_lds_b128 %0, %1, off"
                   :: "v"(l), "v"(gp) : "memory");
      asm volatile("s_wait_asynccnt 0x1" ::: "memory");   // current tile done
    } else {
      asm volatile("s_wait_asynccnt 0x0" ::: "memory");
    }
    const int base = (tile << 10) + tid*4;
    const float4 xv = *(const float4*)(lh + base);
    const float xs[4] = {xv.x, xv.y, xv.z, xv.w};
    #pragma unroll
    for (int k = 0; k < 4; ++k) {
      const int d = clampd(sdur[buf][tid*4 + k]);
      const float ex = expf(xs[k] - M);                    // in (0, 1]
      const unsigned long long fx = (unsigned long long)(ex * SCALE);
      atomicAdd(&A[d], fx);                                // bin total
      if (base + k <= mArr[d]) atomicAdd(&B[d], fx);       // <= last event idx
    }
    buf ^= 1;
  }

  // tail (< 1024 elements): block 0, plain loads
  if (blockIdx.x == 0) {
    for (int idx = (fullTiles << 10) + tid; idx < n; idx += 256) {
      const int d = clampd(dur[idx]);
      const float ex = expf(lh[idx] - M);
      const unsigned long long fx = (unsigned long long)(ex * SCALE);
      atomicAdd(&A[d], fx);
      if (idx <= mArr[d]) atomicAdd(&B[d], fx);
    }
  }
}

// ---------------- finalize: suffix-sum over bins + weighted log, one WGP -----
__global__ void cox_finalize(const unsigned long long* __restrict__ A,
                             const unsigned long long* __restrict__ B,
                             const int* __restrict__ neArr,
                             const unsigned* __restrict__ maxbits,
                             const unsigned* __restrict__ nEv,
                             const unsigned long long* __restrict__ sumLh,
                             float* __restrict__ out)
{
  __shared__ unsigned long long sscan[256];
  __shared__ double sred[256];
  const int t = threadIdx.x;
  const double LN2 = 0.6931471805599453094;
  const float  M   = ord2f(*maxbits);

  // phase 1: Total = sum of all bins (exact integer -> deterministic)
  unsigned long long tot = 0ull;
  for (int i = t; i < MAX_DUR; i += 256) tot += A[i];
  sscan[t] = tot; __syncthreads();
  for (int off = 128; off > 0; off >>= 1) {
    if (t < off) sscan[t] += sscan[t + off];
    __syncthreads();
  }
  const unsigned long long Total = sscan[0];
  __syncthreads();

  // phase 2: chunked inclusive prefix over bins ascending;
  // C_t = Total - P_incl(t) = sum over strictly larger durations.
  double acc = 0.0;
  unsigned long long carry = 0ull;
  for (int basebin = 0; basebin < MAX_DUR; basebin += 256) {
    const int bi = basebin + t;
    unsigned long long a = (bi < MAX_DUR) ? A[bi] : 0ull;
    sscan[t] = a; __syncthreads();
    for (int off = 1; off < 256; off <<= 1) {      // Hillis-Steele u64 scan
      unsigned long long v = (t >= off) ? sscan[t - off] : 0ull;
      __syncthreads();
      sscan[t] += v;
      __syncthreads();
    }
    const unsigned long long incl = carry + sscan[t];
    const unsigned long long chunkSum = sscan[255];
    if (bi < MAX_DUR) {
      const int ne = neArr[bi];
      if (ne > 0) {
        const unsigned long long den = (Total - incl) + B[bi];
        acc += (double)ne *
               ((double)logf((float)den) - 40.0 * LN2 + (double)M);
      }
    }
    carry += chunkSum;
    __syncthreads();
  }

  sred[t] = acc; __syncthreads();
  for (int off = 128; off > 0; off >>= 1) {
    if (t < off) sred[t] += sred[t + off];
    __syncthreads();
  }
  if (t == 0) {
    const double nev = (double)(*nEv);
    const double log_den = sred[0] / nev;
    const double log_num = ((double)(long long)(*sumLh)) / 4294967296.0 / nev;
    out[0] = (float)(log_den - log_num);
  }
}

// ---------------- host-side launcher ----------------------------------------
extern "C" void kernel_launch(void* const* d_in, const int* in_sizes, int n_in,
                              void* d_out, int out_size, void* d_ws, size_t ws_size,
                              hipStream_t stream) {
  (void)n_in; (void)out_size; (void)ws_size;
  const float* lh  = (const float*)d_in[0];
  const int*   dur = (const int*)d_in[1];
  const int*   ev  = (const int*)d_in[2];
  float* out = (float*)d_out;
  const int n = in_sizes[0];

  // workspace layout (2,400,016 bytes total)
  char* ws = (char*)d_ws;
  unsigned long long* A      = (unsigned long long*)(ws);                 // 800000 B
  unsigned long long* B      = (unsigned long long*)(ws + 800000);        // 800000 B
  unsigned long long* sumLh  = (unsigned long long*)(ws + 1600000);       // 8 B
  int*                mArr   = (int*)(ws + 1600008);                      // 400000 B
  int*                neArr  = (int*)(ws + 2000008);                      // 400000 B
  unsigned*           maxbit = (unsigned*)(ws + 2400008);                 // 4 B
  unsigned*           nEv    = (unsigned*)(ws + 2400012);                 // 4 B

  cox_init<<<(MAX_DUR + 255) / 256, 256, 0, stream>>>(A, B, mArr, neArr,
                                                      maxbit, nEv, sumLh);
  cox_pass1<<<1024, 256, 0, stream>>>(lh, dur, ev, n, maxbit, nEv, sumLh,
                                      mArr, neArr);
  cox_pass2<<<2048, 256, 0, stream>>>(lh, dur, n, maxbit, mArr, A, B);
  cox_finalize<<<1, 256, 0, stream>>>(A, B, neArr, maxbit, nEv, sumLh, out);
}
